// MultiScaleAttention_71975061946813
// MI455X (gfx1250) — matrix operand order repaired and checked by reading, not compiled, not run on hardware
//
#include <hip/hip_runtime.h>
#include <hip/hip_bf16.h>

typedef __bf16 bf16_t;
typedef __attribute__((ext_vector_type(16))) __bf16 v16bf;
typedef __attribute__((ext_vector_type(8)))  __bf16 bf16x8;
typedef __attribute__((ext_vector_type(4)))  __bf16 bf16x4;
typedef __attribute__((ext_vector_type(8)))  float   v8f;
typedef __attribute__((ext_vector_type(4)))  float   f32x4;

// ---------------- helpers ----------------
__device__ __forceinline__ v16bf join16(bf16x8 lo, bf16x8 hi) {
  v16bf r;
#pragma unroll
  for (int i = 0; i < 8; ++i) { r[i] = lo[i]; r[i + 8] = hi[i]; }
  return r;
}

__device__ __forceinline__ v8f vzero8() {
  v8f v;
#pragma unroll
  for (int i = 0; i < 8; ++i) v[i] = 0.0f;
  return v;
}

__device__ __forceinline__ v8f wmma_bf16(v16bf a, v16bf b, v8f c) {
  // D(f32 16x16) = A(16x32 bf16) * B(32x16 bf16) + C
  return __builtin_amdgcn_wmma_f32_16x16x32_bf16(false, a, false, b,
                                                 (short)0, c, false, false);
}

// Async global -> LDS copy (16B per lane), tracked by ASYNCcnt.
// LDS aperture sits in addr[63:32], so the low 32 bits of a flat LDS pointer
// are exactly the DS byte offset the instruction wants in its VDST operand.
__device__ __forceinline__ void async_ld_b128(const bf16_t* g, bf16_t* l) {
  unsigned loff = (unsigned)(uintptr_t)l;
  asm volatile("global_load_async_to_lds_b128 %0, %1, off"
               :
               : "v"(loff), "v"(g)
               : "memory");
}

__device__ __forceinline__ void wait_async0() {
  asm volatile("s_wait_asynccnt 0x0" ::: "memory");
}

// ---------------- kernel 0: fp32 -> bf16 convert ----------------
__global__ __launch_bounds__(256) void cvt_f32_to_bf16(
    const float* __restrict__ in, bf16_t* __restrict__ out, int n4) {
  int i = blockIdx.x * 256 + threadIdx.x;
  if (i < n4) {
    f32x4 v = ((const f32x4*)in)[i];
    bf16x4 o;
    o[0] = (bf16_t)v.x; o[1] = (bf16_t)v.y; o[2] = (bf16_t)v.z; o[3] = (bf16_t)v.w;
    ((bf16x4*)out)[i] = o;
  }
}

// ---------------- WMMA GEMM: C[M,N] = A[M,K] * B[N,K]^T + bias ----------------
// 128x128 block tile, 8 waves (4x2), each wave = 32x64 (2x4 WMMA tiles),
// K step 32, double-buffered async global->LDS staging.
template <typename OutT>
__global__ __launch_bounds__(256) void gemm_bf16_kernel(
    const bf16_t* __restrict__ A, const bf16_t* __restrict__ B,
    const float* __restrict__ bias, OutT* __restrict__ C,
    int M, int N, int K) {
  __shared__ bf16_t sA[2][128 * 40];  // 32 cols + pad
  __shared__ bf16_t sB[2][128 * 40];

  const int tid  = threadIdx.x;
  const int lane = tid & 31;
  const int wave = tid >> 5;
  const int wm   = wave & 3;   // 4 waves along M
  const int wn   = wave >> 2;  // 2 waves along N
  const int half = lane >> 4;
  const int l16  = lane & 15;

  const int nbn = N >> 7;
  const int bm  = (blockIdx.x / nbn) << 7;
  const int bn  = (blockIdx.x % nbn) << 7;

  const int ldRow = tid >> 1;         // 0..127
  const int ldCol = (tid & 1) << 4;   // 0 or 16

  v8f acc[2][4];
#pragma unroll
  for (int i = 0; i < 2; ++i)
#pragma unroll
    for (int j = 0; j < 4; ++j) acc[i][j] = vzero8();

  const bf16_t* gA = A + (size_t)(bm + ldRow) * K + ldCol;
  const bf16_t* gB = B + (size_t)(bn + ldRow) * K + ldCol;
  const int lo = ldRow * 40 + ldCol;

  // prologue: stage tile 0 into buffer 0
  async_ld_b128(gA,     &sA[0][lo]);
  async_ld_b128(gA + 8, &sA[0][lo + 8]);
  async_ld_b128(gB,     &sB[0][lo]);
  async_ld_b128(gB + 8, &sB[0][lo + 8]);
  wait_async0();
  __syncthreads();

  const int nIter = K >> 5;
  for (int it = 0; it < nIter; ++it) {
    const int buf = it & 1;
    if (it + 1 < nIter) {  // prefetch next tile into the other buffer
      const int k0 = (it + 1) << 5;
      async_ld_b128(gA + k0,     &sA[buf ^ 1][lo]);
      async_ld_b128(gA + k0 + 8, &sA[buf ^ 1][lo + 8]);
      async_ld_b128(gB + k0,     &sB[buf ^ 1][lo]);
      async_ld_b128(gB + k0 + 8, &sB[buf ^ 1][lo + 8]);
    }

    v16bf bfrag[4];
#pragma unroll
    for (int nt = 0; nt < 4; ++nt) {
      // B frag: n = lane%16, k = e(+8) + 8*half
      const bf16_t* p = &sB[buf][(wn * 64 + nt * 16 + l16) * 40 + half * 8];
      bfrag[nt] = join16(*(const bf16x8*)p, *(const bf16x8*)(p + 16));
    }
#pragma unroll
    for (int sub = 0; sub < 2; ++sub) {
      // A frag: m = lane%16, k = e(+8) + 8*half
      const bf16_t* p = &sA[buf][(wm * 32 + sub * 16 + l16) * 40 + half * 8];
      v16bf afrag = join16(*(const bf16x8*)p, *(const bf16x8*)(p + 16));
#pragma unroll
      for (int nt = 0; nt < 4; ++nt)
        acc[sub][nt] = wmma_bf16(afrag, bfrag[nt], acc[sub][nt]);
    }

    wait_async0();     // prefetched tile has landed
    __syncthreads();   // all waves done reading buf, next tile visible
  }

#pragma unroll
  for (int sub = 0; sub < 2; ++sub) {
#pragma unroll
    for (int nt = 0; nt < 4; ++nt) {
      const int col = bn + wn * 64 + nt * 16 + l16;
      const float bb = bias[col];
#pragma unroll
      for (int r = 0; r < 8; ++r) {
        const int row = bm + wm * 32 + sub * 16 + r + 8 * half;
        C[(size_t)row * N + col] = (OutT)(acc[sub][nt][r] + bb);
      }
    }
  }
}

// ---------------- kernel 2: depthwise pool (k=3, s=2, p=1) ----------------
// qkv: bf16 [B,4096,2304] (n = t*768 + h*64 + d)
// Qp,Kp: bf16 [B,H,2048,64] ; Vt: bf16 [B,H,64,2048] (transposed for PV frags)
__global__ __launch_bounds__(256) void pool_qkv_kernel(
    const bf16_t* __restrict__ qkv,
    const float* __restrict__ wq, const float* __restrict__ wk,
    const float* __restrict__ wv,
    bf16_t* __restrict__ Qp, bf16_t* __restrict__ Kp, bf16_t* __restrict__ Vt) {
  const int idx = blockIdx.x * 256 + threadIdx.x;
  if (idx >= 4 * 12 * 2048 * 64) return;
  const int d  = idx & 63;
  const int lp = (idx >> 6) & 2047;
  const int t  = idx >> 17;          // b*12 + h
  const int h  = t % 12;
  const int b  = t / 12;

  float aq = 0.f, ak = 0.f, av = 0.f;
#pragma unroll
  for (int j = 0; j < 3; ++j) {
    const int l = 2 * lp - 1 + j;
    if (l >= 0 && l < 4096) {
      const size_t base = ((size_t)(b * 4096 + l)) * 2304 + h * 64 + d;
      aq += wq[d * 3 + j] * (float)qkv[base];
      ak += wk[d * 3 + j] * (float)qkv[base + 768];
      av += wv[d * 3 + j] * (float)qkv[base + 1536];
    }
  }
  const size_t o = ((size_t)(t * 2048 + lp)) * 64 + d;
  Qp[o] = (bf16_t)aq;
  Kp[o] = (bf16_t)ak;
  Vt[((size_t)(t * 64 + d)) * 2048 + lp] = (bf16_t)av;
}

// ---------------- kernel 3: flash attention + residual ----------------
// grid = 48 * 16 blocks; block = 256 thr (8 waves); 128 queries / block,
// each wave owns 16 query rows; 64-key blocks double-buffered via async LDS.
__global__ __launch_bounds__(256) void attn_kernel(
    const bf16_t* __restrict__ Qp, const bf16_t* __restrict__ Kp,
    const bf16_t* __restrict__ Vt, bf16_t* __restrict__ Out) {
  __shared__ bf16_t sK[2][64 * 72];   // [key][d]
  __shared__ bf16_t sV[2][64 * 72];   // [d][key]
  __shared__ bf16_t sP[8][16 * 72];   // per-wave P relayout scratch [row][key]

  const int tid  = threadIdx.x;
  const int lane = tid & 31;
  const int wave = tid >> 5;
  const int half = lane >> 4;
  const int l16  = lane & 15;

  const int bh = blockIdx.x >> 4;          // b*12 + h
  const int qb = (blockIdx.x & 15) << 7;   // query block base
  const int b  = bh / 12;
  const int h  = bh % 12;

  const bf16_t* Qbh = Qp + (size_t)bh * 2048 * 64;
  const bf16_t* Kbh = Kp + (size_t)bh * 2048 * 64;
  const bf16_t* Vbh = Vt + (size_t)bh * 64 * 2048;

  const int lrow = tid >> 2;        // 0..63
  const int lcol = (tid & 3) << 4;  // 0,16,32,48
  const int lds_off = lrow * 72 + lcol;

  // prologue: stage key block 0 into buffer 0
  {
    const bf16_t* gk = Kbh + (size_t)lrow * 64 + lcol;
    const bf16_t* gv = Vbh + (size_t)lrow * 2048 + lcol;
    async_ld_b128(gk,     &sK[0][lds_off]);
    async_ld_b128(gk + 8, &sK[0][lds_off + 8]);
    async_ld_b128(gv,     &sV[0][lds_off]);
    async_ld_b128(gv + 8, &sV[0][lds_off + 8]);
  }

  // preload this wave's Q strip as two pre-scaled A fragments (scale = 2^-3, exact in bf16)
  const int qrow = qb + wave * 16 + l16;
  v16bf qf[2];
#pragma unroll
  for (int kk = 0; kk < 2; ++kk) {
    const bf16_t* qp = Qbh + (size_t)qrow * 64 + kk * 32 + half * 8;
    v16bf q = join16(*(const bf16x8*)qp, *(const bf16x8*)(qp + 16));
#pragma unroll
    for (int i = 0; i < 16; ++i) q[i] = (bf16_t)(0.125f * (float)q[i]);
    qf[kk] = q;
  }

  float m_r[8], l_r[8];
#pragma unroll
  for (int r = 0; r < 8; ++r) { m_r[r] = -__builtin_inff(); l_r[r] = 0.0f; }
  v8f acc[4];
#pragma unroll
  for (int i = 0; i < 4; ++i) acc[i] = vzero8();

  wait_async0();
  __syncthreads();

  for (int ib = 0; ib < 32; ++ib) {   // 32 key blocks of 64
    const int buf = ib & 1;
    if (ib + 1 < 32) {  // prefetch next key block into the other buffer
      const int kb = (ib + 1) << 6;
      const bf16_t* gk = Kbh + (size_t)(kb + lrow) * 64 + lcol;
      const bf16_t* gv = Vbh + (size_t)lrow * 2048 + kb + lcol;
      async_ld_b128(gk,     &sK[buf ^ 1][lds_off]);
      async_ld_b128(gk + 8, &sK[buf ^ 1][lds_off + 8]);
      async_ld_b128(gv,     &sV[buf ^ 1][lds_off]);
      async_ld_b128(gv + 8, &sV[buf ^ 1][lds_off + 8]);
    }

    // S = (Q*scale) K^T : 4 key tiles x 2 k-steps over d
    v8f s[4];
#pragma unroll
    for (int nt = 0; nt < 4; ++nt) {
      s[nt] = vzero8();
#pragma unroll
      for (int kk = 0; kk < 2; ++kk) {
        const bf16_t* p = &sK[buf][(nt * 16 + l16) * 72 + kk * 32 + half * 8];
        v16bf bv = join16(*(const bf16x8*)p, *(const bf16x8*)(p + 16));
        s[nt] = wmma_bf16(qf[kk], bv, s[nt]);
      }
    }

    // online softmax (rows live across 16-lane halves)
    float mnew[8], alpha[8];
#pragma unroll
    for (int r = 0; r < 8; ++r) {
      float lm = fmaxf(fmaxf(s[0][r], s[1][r]), fmaxf(s[2][r], s[3][r]));
#pragma unroll
      for (int off = 1; off < 16; off <<= 1) lm = fmaxf(lm, __shfl_xor(lm, off));
      mnew[r]  = fmaxf(m_r[r], lm);
      alpha[r] = __expf(m_r[r] - mnew[r]);
      m_r[r]   = mnew[r];
    }
#pragma unroll
    for (int nt = 0; nt < 4; ++nt)
#pragma unroll
      for (int r = 0; r < 8; ++r) s[nt][r] = __expf(s[nt][r] - mnew[r]);
#pragma unroll
    for (int r = 0; r < 8; ++r) {
      float rs = s[0][r] + s[1][r] + s[2][r] + s[3][r];
#pragma unroll
      for (int off = 1; off < 16; off <<= 1) rs += __shfl_xor(rs, off);
      l_r[r] = l_r[r] * alpha[r] + rs;
#pragma unroll
      for (int dt = 0; dt < 4; ++dt) acc[dt][r] *= alpha[r];
    }

    // relayout P (C-frag -> A-frag) through per-wave LDS scratch
#pragma unroll
    for (int nt = 0; nt < 4; ++nt)
#pragma unroll
      for (int r = 0; r < 8; ++r)
        sP[wave][(r + 8 * half) * 72 + nt * 16 + l16] = (bf16_t)s[nt][r];
    __syncthreads();  // conservative: orders sP store->load (same wave)

    v16bf pf[2];
#pragma unroll
    for (int kk = 0; kk < 2; ++kk) {
      const bf16_t* pp = &sP[wave][l16 * 72 + kk * 32 + half * 8];
      pf[kk] = join16(*(const bf16x8*)pp, *(const bf16x8*)(pp + 16));
    }
    // O += P V : 4 d tiles x 2 k-steps over keys (V^T rows are d)
#pragma unroll
    for (int dt = 0; dt < 4; ++dt)
#pragma unroll
      for (int kk = 0; kk < 2; ++kk) {
        const bf16_t* vp = &sV[buf][(dt * 16 + l16) * 72 + kk * 32 + half * 8];
        v16bf bv = join16(*(const bf16x8*)vp, *(const bf16x8*)(vp + 16));
        acc[dt] = wmma_bf16(pf[kk], bv, acc[dt]);
      }

    wait_async0();     // next K/V block has landed
    __syncthreads();
  }

  // epilogue: normalize, add pooled-Q residual, store [B, L', 768] bf16
#pragma unroll
  for (int dt = 0; dt < 4; ++dt) {
#pragma unroll
    for (int r = 0; r < 8; ++r) {
      const int row = qb + wave * 16 + r + 8 * half;  // lp
      const int d   = dt * 16 + l16;
      const float q = (float)Qbh[(size_t)row * 64 + d];
      const float o = acc[dt][r] / l_r[r] + q;
      Out[((size_t)(b * 2048 + row)) * 768 + h * 64 + d] = (bf16_t)o;
    }
  }
}

// ---------------- host launcher ----------------
extern "C" void kernel_launch(void* const* d_in, const int* in_sizes, int n_in,
                              void* d_out, int out_size, void* d_ws, size_t ws_size,
                              hipStream_t stream) {
  (void)in_sizes; (void)n_in; (void)out_size; (void)ws_size;
  const float* x     = (const float*)d_in[0];
  const float* Wqkv  = (const float*)d_in[1];
  const float* bqkv  = (const float*)d_in[2];
  const float* wq    = (const float*)d_in[3];
  const float* wk    = (const float*)d_in[4];
  const float* wv    = (const float*)d_in[5];
  const float* Wproj = (const float*)d_in[6];
  const float* bproj = (const float*)d_in[7];
  float* out = (float*)d_out;

  char* ws = (char*)d_ws;
  bf16_t* xb     = (bf16_t*)(ws);                // 16384*768*2   = 25,165,824
  bf16_t* wqkvb  = (bf16_t*)(ws + 25165824);     // 2304*768*2   =  3,538,944
  bf16_t* wprojb = (bf16_t*)(ws + 28704768);     // 768*768*2    =  1,179,648
  bf16_t* qkv    = (bf16_t*)(ws + 29884416);     // 16384*2304*2 = 75,497,472
  bf16_t* Qp     = (bf16_t*)(ws + 105381888);    // 48*2048*64*2 = 12,582,912
  bf16_t* Kp     = (bf16_t*)(ws + 117964800);
  bf16_t* Vt     = (bf16_t*)(ws + 130547712);
  bf16_t* att    = (bf16_t*)(ws + 143130624);    // end 155,713,536

  // 0) bf16 conversions
  cvt_f32_to_bf16<<<(16384 * 768 / 4 + 255) / 256, 256, 0, stream>>>(x, xb, 16384 * 768 / 4);
  cvt_f32_to_bf16<<<(2304 * 768 / 4 + 255) / 256, 256, 0, stream>>>(Wqkv, wqkvb, 2304 * 768 / 4);
  cvt_f32_to_bf16<<<(768 * 768 / 4 + 255) / 256, 256, 0, stream>>>(Wproj, wprojb, 768 * 768 / 4);

  // 1) qkv = x @ Wqkv^T + bqkv   (M=16384, N=2304, K=768)
  gemm_bf16_kernel<bf16_t><<<(16384 / 128) * (2304 / 128), 256, 0, stream>>>(
      xb, wqkvb, bqkv, qkv, 16384, 2304, 768);

  // 2) depthwise pooling -> Qp, Kp, Vt
  pool_qkv_kernel<<<(4 * 12 * 2048 * 64) / 256, 256, 0, stream>>>(
      qkv, wq, wk, wv, Qp, Kp, Vt);

  // 3) flash attention + residual -> att [B, 2048, 768] bf16
  attn_kernel<<<48 * 16, 256, 0, stream>>>(Qp, Kp, Vt, att);

  // 4) out = att @ Wproj^T + bproj  (M=8192, N=768, K=768), fp32 output
  gemm_bf16_kernel<float><<<(8192 / 128) * (768 / 128), 256, 0, stream>>>(
      att, wprojb, bproj, out, 8192, 768, 768);
}
